// MLP_24850680774784
// MI455X (gfx1250) — compile-verified, matrix operand order checked by reading, not run on hardware
//
#include <hip/hip_runtime.h>

// ---------------------------------------------------------------------------
// MLP (128->256->128->2, tanh) + softmax + segment-mean for MI455X (gfx1250).
// f16 WMMA with f32 accumulate; weights pre-swizzled into B-fragment layout.
// Each wave processes TWO 16-row tiles so every B fragment fetched from L2
// feeds two WMMAs (halves weight traffic). Activations use v_tanh_f32.
// ---------------------------------------------------------------------------

typedef __attribute__((ext_vector_type(16))) _Float16 v16h;
typedef __attribute__((ext_vector_type(8)))  _Float16 v8h;
typedef __attribute__((ext_vector_type(8)))  float    v8f;

#define D_IN   128
#define H1_DIM 256
#define H2_DIM 128
#define WPG    2            // waves per workgroup (64 threads)
#define S1     264          // LDS f16 row stride for H1 tile buffer (16B aligned, bank-skewed)
#define S2     136          // LDS f16 row stride for H2 tile buffer

__device__ __forceinline__ v8f wmma_f16(v16h a, v16h b, v8f c) {
  // 8 args: (neg_a, A, neg_b, B, c_mod, C, reuse_a, reuse_b)
  return __builtin_amdgcn_wmma_f32_16x16x32_f16(false, a, false, b, (short)0, c,
                                                false, false);
}

__device__ __forceinline__ float fast_tanh(float x) {
#if __has_builtin(__builtin_amdgcn_tanhf)
  return __builtin_amdgcn_tanhf(x);            // v_tanh_f32 (TRANS, co-executes w/ WMMA)
#else
  float e = __expf(2.0f * x);
  return 1.0f - 2.0f * __builtin_amdgcn_rcpf(e + 1.0f);
#endif
}

__device__ __forceinline__ float fast_sigmoid(float x) {  // 1/(1+e^x)
  return __builtin_amdgcn_rcpf(1.0f + __expf(x));
}

// A-fragment gather from an LDS tile buffer (two bank-skewed ds_load_b128)
__device__ __forceinline__ v16h lds_afrag(const _Float16* buf, int stride, int c,
                                          int nl, int hl) {
  const v8h* lo = (const v8h*)&buf[nl * stride + 32 * c + 8 * hl];
  const v8h* hi = (const v8h*)&buf[nl * stride + 32 * c + 16 + 8 * hl];
  return __builtin_shufflevector(*lo, *hi, 0, 1, 2, 3, 4, 5, 6, 7, 8, 9, 10, 11,
                                 12, 13, 14, 15);
}

__global__ void zero_kernel(float* __restrict__ p, int n) {
  int i = blockIdx.x * blockDim.x + threadIdx.x;
  if (i < n) p[i] = 0.0f;
}

// Build f16 B-fragments for W1 (64 frags), W2 (64 frags), W3 padded to 16 cols (4 frags).
// Fragment (c,j): lane l holds column n = 16j + (l&15); element e holds
// k = 32c + 16*(e>=8) + 8*(l>>4) + (e&7)   (mirror of the documented 16-bit A layout).
__global__ void prep_weights(const float* __restrict__ W1, const float* __restrict__ W2,
                             const float* __restrict__ W3,
                             v16h* __restrict__ w1f, v16h* __restrict__ w2f,
                             v16h* __restrict__ w3f) {
  int tid  = blockIdx.x * blockDim.x + threadIdx.x;
  int lane = tid & 31;
  int f    = tid >> 5;
  int nl = lane & 15;
  int hl = lane >> 4;
  if (f < 64) {                       // W1: c = f>>4 (k-chunk of 32), j = f&15 (col tile)
    int c = f >> 4, j = f & 15;
    v16h v;
#pragma unroll
    for (int e = 0; e < 16; ++e) {
      int k = 32 * c + ((e & 8) ? 16 : 0) + 8 * hl + (e & 7);
      v[e] = (_Float16)W1[k * H1_DIM + 16 * j + nl];
    }
    w1f[f * 32 + lane] = v;
  } else if (f < 128) {               // W2: c = (f-64)>>3, j = (f-64)&7
    int f2 = f - 64;
    int c = f2 >> 3, j = f2 & 7;
    v16h v;
#pragma unroll
    for (int e = 0; e < 16; ++e) {
      int k = 32 * c + ((e & 8) ? 16 : 0) + 8 * hl + (e & 7);
      v[e] = (_Float16)W2[k * H2_DIM + 16 * j + nl];
    }
    w2f[f2 * 32 + lane] = v;
  } else if (f < 132) {               // W3 (128x2) zero-padded to 128x16
    int c = f - 128;
    v16h v;
#pragma unroll
    for (int e = 0; e < 16; ++e) {
      int k = 32 * c + ((e & 8) ? 16 : 0) + 8 * hl + (e & 7);
      v[e] = (nl < 2) ? (_Float16)W3[k * 2 + nl] : (_Float16)0.0f;
    }
    w3f[c * 32 + lane] = v;
  }
}

__global__ __launch_bounds__(WPG * 32) void mlp_seg_kernel(
    const float* __restrict__ x,
    const float* __restrict__ b1, const float* __restrict__ b2,
    const float* __restrict__ b3,
    const v16h* __restrict__ w1f, const v16h* __restrict__ w2f,
    const v16h* __restrict__ w3f,
    float* __restrict__ sums, float* __restrict__ counts,
    int nrows, int ntiles) {
  __shared__ _Float16 sh1[WPG][2][16 * S1];   // per-wave, per-tile H1 buffers
  __shared__ _Float16 sh2[WPG][2][16 * S2];   // per-wave, per-tile H2 buffers

  const int lane = threadIdx.x & 31;
  const int wave = threadIdx.x >> 5;
  const int nl   = lane & 15;   // A: row-in-tile / B,C,D: column
  const int hl   = lane >> 4;   // lane-half

  const int pair = blockIdx.x * WPG + wave;
  int tl[2];
  bool act[2];
  tl[0]  = pair * 2;
  tl[1]  = pair * 2 + 1;
  act[0] = tl[0] < ntiles;
  act[1] = tl[1] < ntiles;
  if (!act[0]) tl[0] = 0;                    // keep control flow uniform for WMMA
  if (!act[1]) tl[1] = tl[0];

  // ---- X -> f16 A-fragments for both tiles (dword loads; row stride 129 f32)
  v16h A1[2][4];
#pragma unroll
  for (int t = 0; t < 2; ++t) {
    int rsrc = tl[t] * 16 + nl;
    if (rsrc >= nrows) rsrc = nrows - 1;
    const float* xrow = x + (size_t)rsrc * (D_IN + 1);
#pragma unroll
    for (int c = 0; c < 4; ++c) {
      v16h a;
#pragma unroll
      for (int e = 0; e < 16; ++e) {
        int k = 32 * c + ((e & 8) ? 16 : 0) + 8 * hl + (e & 7);
        a[e] = (_Float16)xrow[k];
      }
      A1[t][c] = a;
    }
  }

  // ---- layer 1: each W1 fragment fetched once, used for both tiles
  _Float16* h1a = sh1[wave][0];
  _Float16* h1b = sh1[wave][1];
  for (int j = 0; j < 16; ++j) {
    v8f acc0 = {}, acc1 = {};
#pragma unroll
    for (int c = 0; c < 4; ++c) {
      v16h b = w1f[(c * 16 + j) * 32 + lane];
      acc0 = wmma_f16(A1[0][c], b, acc0);
      acc1 = wmma_f16(A1[1][c], b, acc1);
    }
    float bias = b1[16 * j + nl];
#pragma unroll
    for (int r = 0; r < 8; ++r) {      // C/D layout: row = r + 8*hl, col = nl
      h1a[(r + 8 * hl) * S1 + 16 * j + nl] = (_Float16)fast_tanh(acc0[r] + bias);
      h1b[(r + 8 * hl) * S1 + 16 * j + nl] = (_Float16)fast_tanh(acc1[r] + bias);
    }
  }
  __syncthreads();

  // ---- layer 2: A-fragments streamed from LDS on demand; W2 frag shared by tiles
  _Float16* h2a = sh2[wave][0];
  _Float16* h2b = sh2[wave][1];
  for (int j = 0; j < 8; ++j) {
    v8f acc0 = {}, acc1 = {};
#pragma unroll
    for (int c = 0; c < 8; ++c) {
      v16h b = w2f[(c * 8 + j) * 32 + lane];
      acc0 = wmma_f16(lds_afrag(h1a, S1, c, nl, hl), b, acc0);
      acc1 = wmma_f16(lds_afrag(h1b, S1, c, nl, hl), b, acc1);
    }
    float bias = b2[16 * j + nl];
#pragma unroll
    for (int r = 0; r < 8; ++r) {
      h2a[(r + 8 * hl) * S2 + 16 * j + nl] = (_Float16)fast_tanh(acc0[r] + bias);
      h2b[(r + 8 * hl) * S2 + 16 * j + nl] = (_Float16)fast_tanh(acc1[r] + bias);
    }
  }
  __syncthreads();

  // ---- layer 3: logits (cols 0..1 valid) = H2 @ W3pad, both tiles
  v8f acc3[2] = {{}, {}};
#pragma unroll
  for (int c = 0; c < 4; ++c) {
    v16h b = w3f[c * 32 + lane];
    acc3[0] = wmma_f16(lds_afrag(h2a, S2, c, nl, hl), b, acc3[0]);
    acc3[1] = wmma_f16(lds_afrag(h2b, S2, c, nl, hl), b, acc3[1]);
  }
  float bias3 = (nl < 2) ? b3[nl] : 0.0f;
  float tv[2][8], ot[2][8];
#pragma unroll
  for (int t = 0; t < 2; ++t) {
#pragma unroll
    for (int r = 0; r < 8; ++r) tv[t][r] = fast_tanh(acc3[t][r] + bias3);
#pragma unroll
    for (int r = 0; r < 8; ++r) ot[t][r] = __shfl_xor(tv[t][r], 1, 32);
  }

  // ---- 2-class softmax + segment accumulation (lanes holding column 0)
  if (nl == 0) {
#pragma unroll
    for (int t = 0; t < 2; ++t) {
      if (!act[t]) continue;
#pragma unroll
      for (int r = 0; r < 8; ++r) {
        int rr = tl[t] * 16 + r + 8 * hl;
        if (rr < nrows) {
          float p0 = fast_sigmoid(ot[t][r] - tv[t][r]);   // 1/(1+e^(t1-t0))
          int g = (int)x[(size_t)rr * (D_IN + 1) + D_IN];
          atomicAdd(&sums[2 * g], p0);
          atomicAdd(&sums[2 * g + 1], 1.0f - p0);
          atomicAdd(&counts[g], 1.0f);
        }
      }
    }
  }
}

__global__ void finalize_kernel(const float* __restrict__ sums,
                                const float* __restrict__ counts,
                                float* __restrict__ out, int G) {
  int g = blockIdx.x * blockDim.x + threadIdx.x;
  if (g < G) {
    float c = counts[g];
    float d = (c > 0.0f) ? c : 1.0f;
    out[2 * g]     = sums[2 * g] / d;
    out[2 * g + 1] = sums[2 * g + 1] / d;
  }
}

extern "C" void kernel_launch(void* const* d_in, const int* in_sizes, int n_in,
                              void* d_out, int out_size, void* d_ws, size_t ws_size,
                              hipStream_t stream) {
  const float* x  = (const float*)d_in[0];
  const float* W1 = (const float*)d_in[1];
  const float* b1 = (const float*)d_in[2];
  const float* W2 = (const float*)d_in[3];
  const float* b2 = (const float*)d_in[4];
  const float* W3 = (const float*)d_in[5];
  const float* b3 = (const float*)d_in[6];

  const int nrows = in_sizes[0] / (D_IN + 1);
  const int G     = out_size / 2;

  char* ws = (char*)d_ws;
  v16h* w1f = (v16h*)(ws);                 // 64 frags * 1KB = 64 KB
  v16h* w2f = (v16h*)(ws + 65536);         // 64 KB
  v16h* w3f = (v16h*)(ws + 131072);        // 4 KB
  float* sums   = (float*)(ws + 135168);                   // 2*G f32
  float* counts = (float*)(ws + 135168 + (size_t)8 * G);   // G f32 (contiguous)

  // zero accumulators every call (deterministic; harness doesn't re-poison)
  int nz = 3 * G;
  zero_kernel<<<(nz + 255) / 256, 256, 0, stream>>>(sums, nz);

  prep_weights<<<(132 * 32 + 255) / 256, 256, 0, stream>>>(W1, W2, W3, w1f, w2f, w3f);

  int ntiles = (nrows + 15) / 16;
  int npairs = (ntiles + 1) / 2;
  int nwg    = (npairs + WPG - 1) / WPG;
  mlp_seg_kernel<<<nwg, WPG * 32, 0, stream>>>(x, b1, b2, b3, w1f, w2f, w3f,
                                               sums, counts, nrows, ntiles);

  finalize_kernel<<<(G + 255) / 256, 256, 0, stream>>>(sums, counts, (float*)d_out, G);
}
